// LDPC_40896678592635
// MI455X (gfx1250) — 8-bit matrix operand order repaired and checked by reading, compiled, not run on hardware
//
#include <hip/hip_runtime.h>
#include <math.h>


#define NVAR   96
#define NCHK   48
#define DC     6
#define NEDGE  288
#define KOUT   48
#define TB     64
#define NITER  96
#define LLR_SCALE 2.1972245773362196f
#define EPSV   1e-7f

typedef __attribute__((ext_vector_type(8))) int v8i;

__global__ __launch_bounds__(TB) void ldpc_bp_kernel(
    const float* __restrict__ recv,
    const int*   __restrict__ col_idx,
    float*       __restrict__ out,
    int Btot)
{
    __shared__ float s_mvc[NEDGE * TB];
    __shared__ float s_tot[NVAR  * TB];
    __shared__ float s_llr[NVAR  * TB];
    __shared__ int   s_col[NEDGE];

    const int t = threadIdx.x;
    const int b = blockIdx.x * TB + t;

    for (int e = t; e < NEDGE; e += TB) s_col[e] = col_idx[e];
    __syncthreads();

    for (int n = 0; n < NVAR; ++n) {
        float r = recv[(size_t)b * NVAR + n];
        s_llr[n * TB + t] = LLR_SCALE * (1.0f - 2.0f * r);
    }
    for (int e = 0; e < NEDGE; ++e)
        s_mvc[e * TB + t] = s_llr[s_col[e] * TB + t];

    for (int it = 0; it < NITER; ++it) {
        for (int n = 0; n < NVAR; ++n)
            s_tot[n * TB + t] = s_llr[n * TB + t];

        for (int m = 0; m < NCHK; ++m) {
            float lg[DC];
            int   ng[DC];
            float slog = 0.0f;
            int   sneg = 0;
            #pragma unroll
            for (int j = 0; j < DC; ++j) {
                float mv = s_mvc[(DC * m + j) * TB + t];
                float tt = tanhf(0.5f * mv);
                float a  = fabsf(tt);
                a = fminf(fmaxf(a, EPSV), 1.0f - EPSV);
                float l  = __logf(a);
                int   s  = (tt < 0.0f) ? 1 : 0;
                lg[j] = l; ng[j] = s;
                slog += l; sneg += s;
            }
            #pragma unroll
            for (int j = 0; j < DC; ++j) {
                float em  = __expf(slog - lg[j]);
                int   par = (sneg - ng[j]) & 1;
                float ext = par ? -em : em;
                ext = fminf(fmaxf(ext, -1.0f + EPSV), 1.0f - EPSV);
                float mcv = __logf((1.0f + ext) / (1.0f - ext));
                int e = DC * m + j;
                s_mvc[e * TB + t] = mcv;
                s_tot[s_col[e] * TB + t] += mcv;
            }
        }
        for (int e = 0; e < NEDGE; ++e) {
            float mcv = s_mvc[e * TB + t];
            s_mvc[e * TB + t] = s_tot[s_col[e] * TB + t] - mcv;
        }
    }

    float* out_full = out + (size_t)Btot * KOUT;
    for (int n = 0; n < NVAR; ++n) {
        float bit = (s_tot[n * TB + t] < 0.0f) ? 1.0f : 0.0f;
        out_full[(size_t)b * NVAR + n] = bit;
        if (n < KOUT) out[(size_t)b * KOUT + n] = bit;
    }
}

__global__ __launch_bounds__(256) void ldpc_syndrome_wmma_kernel(
    const float* __restrict__ corrected,
    const int*   __restrict__ H,
    float*       __restrict__ valid,
    int Btot)
{
    const int wave = (blockIdx.x * blockDim.x + threadIdx.x) >> 5;
    const int lane = threadIdx.x & 31;
    const int b0   = wave * 16;
    if (b0 >= Btot) return;

    const int half = (lane >= 16) ? 1 : 0;
    const int l15  = lane & 15;

    v8i a[2];
    #pragma unroll
    for (int kt = 0; kt < 2; ++kt) {
        #pragma unroll
        for (int v = 0; v < 8; ++v) {
            int baseK = ((v >> 1) << 4) + ((v & 1) << 2) + (half << 3);
            unsigned w = 0u;
            #pragma unroll
            for (int byte = 0; byte < 4; ++byte) {
                int k = kt * 64 + baseK + byte;
                unsigned bit = 0u;
                if (k < NVAR)
                    bit = (corrected[(size_t)(b0 + l15) * NVAR + k] != 0.0f) ? 1u : 0u;
                w |= bit << (8 * byte);
            }
            a[kt][v] = (int)w;
        }
    }

    unsigned badmask = 0u;
    #pragma unroll
    for (int nt = 0; nt < 3; ++nt) {
        v8i c = {};
        #pragma unroll
        for (int kt = 0; kt < 2; ++kt) {
            v8i bm;
            #pragma unroll
            for (int v = 0; v < 8; ++v) {
                unsigned w = 0u;
                #pragma unroll
                for (int byte = 0; byte < 4; ++byte) {
                    int k = kt * 64 + ((v >> 1) << 4) + ((v & 1) << 2) + (half << 3) + byte;
                    int m = nt * 16 + l15;
                    unsigned bit = 0u;
                    if (k < NVAR) bit = (unsigned)(H[m * NVAR + k] & 1);
                    w |= bit << (8 * byte);
                }
                bm[v] = (int)w;
            }
            c = __builtin_amdgcn_wmma_i32_16x16x64_iu8(
                     false, a[kt],  false, bm, c,
                     false,  false);
        }
        #pragma unroll
        for (int r = 0; r < 8; ++r)
            badmask |= ((unsigned)c[r] & 1u) << r;
    }

    #pragma unroll
    for (int off = 1; off < 16; off <<= 1)
        badmask |= (unsigned)__shfl_xor((int)badmask, off, 16);

    if (l15 < 8) {
        int row = l15 + half * 8;
        unsigned bad = (badmask >> l15) & 1u;
        valid[b0 + row] = bad ? 0.0f : 1.0f;
    }
}

extern "C" void kernel_launch(void* const* d_in, const int* in_sizes, int n_in,
                              void* d_out, int out_size, void* d_ws, size_t ws_size,
                              hipStream_t stream) {
    (void)n_in; (void)out_size; (void)d_ws; (void)ws_size;

    const float* recv    = (const float*)d_in[0];
    const int*   H       = (const int*)d_in[1];
    const int*   col_idx = (const int*)d_in[3];
    const int    Btot    = in_sizes[0] / NVAR;

    float* out = (float*)d_out;

    ldpc_bp_kernel<<<Btot / TB, TB, 0, stream>>>(recv, col_idx, out, Btot);

    const float* corrected = out + (size_t)Btot * KOUT;
    float*       valid     = out + (size_t)Btot * (KOUT + NVAR);
    const int waves   = Btot / 16;
    const int threads = 256;
    const int blocks  = (waves * 32 + threads - 1) / threads;
    ldpc_syndrome_wmma_kernel<<<blocks, threads, 0, stream>>>(corrected, H, valid, Btot);
}
